// SplineNet_7876970020892
// MI455X (gfx1250) — matrix operand order repaired and checked by reading, not compiled, not run on hardware
//
#include <hip/hip_runtime.h>
#include <hip/hip_bf16.h>

typedef __attribute__((ext_vector_type(16))) _Float16 v16h;
typedef __attribute__((ext_vector_type(8)))  float    v8f;

#define FIN 256

// ---------------------------------------------------------------------------
// Layer-1 node GEMM: Y1[N,48] = x[N,256] @ [W1[0] | W1[1] | root1]  (f16 WMMA)
// One wave computes a 16-row tile; 3 column tiles of 16; K = 8 steps of 32.
// ---------------------------------------------------------------------------
__global__ void gemm1_wmma(const float* __restrict__ x,
                           const float* __restrict__ W1,
                           const float* __restrict__ root1,
                           float* __restrict__ Y1, int N) {
  // fragment-major weight staging: [ktile(8)][t(3)][lane(32)][j(16)] halves
  __shared__ alignas(32) _Float16 ldsB[8 * 3 * 32 * 16];  // 24 KB
  const int tid = threadIdx.x;
  for (int i = tid; i < 8 * 3 * 32 * 16; i += blockDim.x) {
    int j    = i & 15;
    int lane = (i >> 4) & 31;
    int tt   = (i >> 9) % 3;
    int kt   = (i >> 9) / 3;
    int col  = lane & 15;
    int k    = kt * 32 + ((lane >> 4) * 8) + (j & 7) + ((j >> 3) * 16);   // B-frag: K=(lane/16)*8+(j&7)+(j/8)*16
    const float* B = (tt == 0) ? W1 : (tt == 1) ? (W1 + 256 * 16) : root1;
    ldsB[i] = (_Float16)B[k * 16 + col];
  }
  __syncthreads();

  const int wave  = tid >> 5;
  const int lane  = tid & 31;
  const int tileM = blockIdx.x * (blockDim.x >> 5) + wave;
  const int row0  = tileM * 16;
  if (row0 >= N) return;                 // wave-uniform exit, EXEC stays full
  int row = row0 + (lane & 15);
  if (row >= N) row = N - 1;             // clamp (select), keep EXEC all-ones
  const int kb = (lane >> 4) * 8;        // A-frag: K = kb+j and kb+16+j
  const float* xrow = x + (size_t)row * FIN + kb;

  v8f acc0 = {}, acc1 = {}, acc2 = {};
  for (int kt = 0; kt < 8; ++kt) {
    const int k0 = kt * 32;
    v16h a;
#pragma unroll
    for (int j = 0; j < 8; ++j) {
      a[j]     = (_Float16)xrow[k0 + j];
      a[j + 8] = (_Float16)xrow[k0 + 16 + j];
    }
    v16h b0 = *(const v16h*)&ldsB[((kt * 3 + 0) * 32 + lane) * 16];
    v16h b1 = *(const v16h*)&ldsB[((kt * 3 + 1) * 32 + lane) * 16];
    v16h b2 = *(const v16h*)&ldsB[((kt * 3 + 2) * 32 + lane) * 16];
    acc0 = __builtin_amdgcn_wmma_f32_16x16x32_f16(false, a, false, b0, (short)0, acc0, false, false);
    acc1 = __builtin_amdgcn_wmma_f32_16x16x32_f16(false, a, false, b1, (short)0, acc1, false, false);
    acc2 = __builtin_amdgcn_wmma_f32_16x16x32_f16(false, a, false, b2, (short)0, acc2, false, false);
  }
  // D layout: VGPR v, lane L -> row = v + 8*(L>>4), col = L&15
  const int col   = lane & 15;
  const int rbase = row0 + ((lane >> 4) << 3);
#pragma unroll
  for (int v = 0; v < 8; ++v) {
    int r = rbase + v;
    if (r < N) {
      float* yr = Y1 + (size_t)r * 48;
      yr[col]      = acc0[v];
      yr[16 + col] = acc1[v];
      yr[32 + col] = acc2[v];
    }
  }
}

// ---------------------------------------------------------------------------
// Layer-2 node GEMM: Y2[N,96] = h[N,16] @ [W2[0] | W2[1] | root2] (K padded 32)
// ---------------------------------------------------------------------------
__global__ void gemm2_wmma(const float* __restrict__ h,
                           const float* __restrict__ W2,
                           const float* __restrict__ root2,
                           float* __restrict__ Y2, int N) {
  __shared__ alignas(32) _Float16 ldsB[6 * 32 * 16];  // 6 KB
  const int tid = threadIdx.x;
  for (int i = tid; i < 6 * 32 * 16; i += blockDim.x) {
    int j    = i & 15;
    int lane = (i >> 4) & 31;
    int tt   = i >> 9;
    int k    = ((lane >> 4) * 8) + (j & 7) + ((j >> 3) * 16);   // 0..31 (k>=16 is zero pad)
    int c    = tt * 16 + (lane & 15);    // 0..95
    float v = 0.f;
    if (k < 16) {
      if (c < 64) v = W2[(size_t)(c >> 5) * 16 * 32 + k * 32 + (c & 31)];
      else        v = root2[k * 32 + (c - 64)];
    }
    ldsB[i] = (_Float16)v;
  }
  __syncthreads();

  const int wave  = tid >> 5;
  const int lane  = tid & 31;
  const int tileM = blockIdx.x * (blockDim.x >> 5) + wave;
  const int row0  = tileM * 16;
  if (row0 >= N) return;
  int row = row0 + (lane & 15);
  if (row >= N) row = N - 1;
  const int kb = (lane >> 4) * 8;
  const float* hr = h + (size_t)row * 16 + kb;

  v16h a;
#pragma unroll
  for (int j = 0; j < 8; ++j) { a[j] = (_Float16)hr[j]; a[j + 8] = (_Float16)0.f; }

  v8f acc[6];
#pragma unroll
  for (int t = 0; t < 6; ++t) {
    v16h b = *(const v16h*)&ldsB[(t * 32 + lane) * 16];
    v8f c = {};
    acc[t] = __builtin_amdgcn_wmma_f32_16x16x32_f16(false, a, false, b, (short)0, c, false, false);
  }
  const int col   = lane & 15;
  const int rbase = row0 + ((lane >> 4) << 3);
#pragma unroll
  for (int v = 0; v < 8; ++v) {
    int r = rbase + v;
    if (r < N) {
      float* yr = Y2 + (size_t)r * 96;
#pragma unroll
      for (int t = 0; t < 6; ++t) yr[t * 16 + col] = acc[t][v];
    }
  }
}

// ---------------------------------------------------------------------------
// Edge scatter layer 1: 16 feats/edge. msg = (1-u)*xs0[src] + u*xs1[src]
// ---------------------------------------------------------------------------
__global__ void scatter1(const float* __restrict__ Y1, const int* __restrict__ src,
                         const int* __restrict__ dst, const float* __restrict__ ea,
                         float* __restrict__ agg1, float* __restrict__ cnt,
                         long long E) {
  long long gid = (long long)blockIdx.x * blockDim.x + threadIdx.x;
  long long e = gid >> 4;
  if (e >= E) return;
  int f = (int)(gid & 15);
  int s = src[e], d = dst[e];
  float u = fminf(fmaxf(ea[e], 0.f), 1.f);
  const float* ys = Y1 + (size_t)s * 48;
  float m = (1.f - u) * ys[f] + u * ys[16 + f];
  atomicAdd(&agg1[(size_t)d * 16 + f], m);
  if (f == 0) atomicAdd(&cnt[d], 1.f);
}

// ---------------------------------------------------------------------------
// Edge scatter layer 2: 32 feats/edge.
// ---------------------------------------------------------------------------
__global__ void scatter2(const float* __restrict__ Y2, const int* __restrict__ src,
                         const int* __restrict__ dst, const float* __restrict__ ea,
                         float* __restrict__ agg2, long long E) {
  long long gid = (long long)blockIdx.x * blockDim.x + threadIdx.x;
  long long e = gid >> 5;
  if (e >= E) return;
  int f = (int)(gid & 31);
  int s = src[e], d = dst[e];
  float u = fminf(fmaxf(ea[e], 0.f), 1.f);
  const float* ys = Y2 + (size_t)s * 96;
  float m = (1.f - u) * ys[f] + u * ys[32 + f];
  atomicAdd(&agg2[(size_t)d * 32 + f], m);
}

// ---------------------------------------------------------------------------
// Finalize layer 1: h = elu(agg/max(cnt,1) + x@root1 + bias1)
// ---------------------------------------------------------------------------
__global__ void finalize1(const float* __restrict__ agg1, const float* __restrict__ cnt,
                          const float* __restrict__ Y1, const float* __restrict__ bias1,
                          float* __restrict__ h, int N) {
  int gid = blockIdx.x * blockDim.x + threadIdx.x;
  int n = gid >> 4;
  if (n >= N) return;
  int f = gid & 15;
  float c = fmaxf(cnt[n], 1.f);
  float v = agg1[(size_t)n * 16 + f] / c + Y1[(size_t)n * 48 + 32 + f] + bias1[f];
  h[(size_t)n * 16 + f] = (v > 0.f) ? v : (expf(v) - 1.f);
}

// ---------------------------------------------------------------------------
// Finalize layer 2 + log_softmax. One wave32 per node; lane = class.
// ---------------------------------------------------------------------------
__global__ void finalize2(const float* __restrict__ agg2, const float* __restrict__ cnt,
                          const float* __restrict__ Y2, const float* __restrict__ bias2,
                          float* __restrict__ out, int N) {
  int tid = threadIdx.x;
  int n = blockIdx.x * (blockDim.x >> 5) + (tid >> 5);
  if (n >= N) return;
  int c = tid & 31;
  float deg = fmaxf(cnt[n], 1.f);
  float v = agg2[(size_t)n * 32 + c] / deg + Y2[(size_t)n * 96 + 64 + c] + bias2[c];
  float m = v;
#pragma unroll
  for (int off = 16; off > 0; off >>= 1) m = fmaxf(m, __shfl_xor(m, off, 32));
  float ex = expf(v - m);
  float s = ex;
#pragma unroll
  for (int off = 16; off > 0; off >>= 1) s += __shfl_xor(s, off, 32);
  out[(size_t)n * 32 + c] = v - m - logf(s);
}

extern "C" void kernel_launch(void* const* d_in, const int* in_sizes, int n_in,
                              void* d_out, int out_size, void* d_ws, size_t ws_size,
                              hipStream_t stream) {
  (void)n_in; (void)out_size; (void)ws_size;
  const float* x     = (const float*)d_in[0];
  const float* ea    = (const float*)d_in[1];
  const float* W1    = (const float*)d_in[2];
  const float* root1 = (const float*)d_in[3];
  const float* bias1 = (const float*)d_in[4];
  const float* W2    = (const float*)d_in[5];
  const float* root2 = (const float*)d_in[6];
  const float* bias2 = (const float*)d_in[7];
  const int*   ei    = (const int*)d_in[8];

  const int N = in_sizes[0] / FIN;
  const long long E = in_sizes[8] / 2;
  const int* srcp = ei;
  const int* dstp = ei + E;

  char* ws = (char*)d_ws;
  size_t off = 0;
  auto alloc = [&](size_t bytes) -> void* {
    void* p = ws + off;
    off += (bytes + 255) & ~(size_t)255;
    return p;
  };
  float* Y1   = (float*)alloc((size_t)N * 48 * sizeof(float));
  float* agg1 = (float*)alloc((size_t)N * 16 * sizeof(float));
  float* cnt  = (float*)alloc((size_t)N * sizeof(float));
  float* h    = (float*)alloc((size_t)N * 16 * sizeof(float));
  float* Y2   = (float*)alloc((size_t)N * 96 * sizeof(float));
  float* agg2 = (float*)alloc((size_t)N * 32 * sizeof(float));

  hipMemsetAsync(agg1, 0, (size_t)N * 16 * sizeof(float), stream);
  hipMemsetAsync(cnt,  0, (size_t)N * sizeof(float), stream);
  hipMemsetAsync(agg2, 0, (size_t)N * 32 * sizeof(float), stream);

  const int wavesPerBlock = 4;               // 128 threads = 4 wave32
  const int tiles = (N + 15) / 16;
  const int gGemm = (tiles + wavesPerBlock - 1) / wavesPerBlock;

  gemm1_wmma<<<gGemm, 128, 0, stream>>>(x, W1, root1, Y1, N);

  long long t1 = E * 16;
  scatter1<<<(int)((t1 + 255) / 256), 256, 0, stream>>>(Y1, srcp, dstp, ea, agg1, cnt, E);

  finalize1<<<((N * 16) + 255) / 256, 256, 0, stream>>>(agg1, cnt, Y1, bias1, h, N);

  gemm2_wmma<<<gGemm, 128, 0, stream>>>(h, W2, root2, Y2, N);

  long long t2 = E * 32;
  scatter2<<<(int)((t2 + 255) / 256), 256, 0, stream>>>(Y2, srcp, dstp, ea, agg2, E);

  finalize2<<<(N + 7) / 8, 256, 0, stream>>>(agg2, cnt, Y2, bias2, (float*)d_out, N);
}
